// _PackedExpertsMoE_21938692948540
// MI455X (gfx1250) — matrix operand order repaired and checked by reading, not compiled, not run on hardware
//
#include <hip/hip_runtime.h>
#include <math.h>

#define H_DIM   1024
#define FF_DIM  4096
#define E_NUM   8
#define T_NUM   16384
#define CAP     2048
#define RMS_EPS 1e-5f

// ---------------------------------------------------------------- types
typedef __attribute__((ext_vector_type(16))) __bf16 v16bf;
typedef __attribute__((ext_vector_type(8)))  float  v8f;
typedef __attribute__((ext_vector_type(4)))  float  f4v;

union ABFrag { v16bf v; f4v f[2]; };

__device__ __forceinline__ unsigned short f2bf(float f) {
  unsigned int u = __float_as_uint(f);
  u += 0x7FFFu + ((u >> 16) & 1u);            // round-to-nearest-even
  return (unsigned short)(u >> 16);
}

__device__ __forceinline__ float gelu_exact(float v) {
  return 0.5f * v * (1.0f + erff(v * 0.70710678118654752440f));
}

// ---------------------------------------------------------------- f32 -> bf16
__global__ void k_f32_to_bf16x4(const float4* __restrict__ in,
                                ushort4* __restrict__ out, int n4) {
  int i = blockIdx.x * blockDim.x + threadIdx.x;
  int stride = gridDim.x * blockDim.x;
  for (; i < n4; i += stride) {
    float4 v = in[i];
    ushort4 o;
    o.x = f2bf(v.x); o.y = f2bf(v.y); o.z = f2bf(v.z); o.w = f2bf(v.w);
    out[i] = o;
  }
}

// ---------------------------------------------------------------- RMSNorm + router + softmax
__global__ __launch_bounds__(256)
void k_rmsnorm_router(const float* __restrict__ x, const float* __restrict__ nw,
                      const float* __restrict__ rw, const float* __restrict__ rb,
                      unsigned short* __restrict__ xn, float* __restrict__ probsT) {
  const int t = blockIdx.x, tid = threadIdx.x;
  __shared__ float red[256];
  __shared__ float logits[E_NUM];
  __shared__ float sscale;

  float4 v = ((const float4*)(x + (size_t)t * H_DIM))[tid];
  red[tid] = v.x * v.x + v.y * v.y + v.z * v.z + v.w * v.w;
  __syncthreads();
  for (int s = 128; s > 0; s >>= 1) {
    if (tid < s) red[tid] += red[tid + s];
    __syncthreads();
  }
  if (tid == 0) sscale = rsqrtf(red[0] * (1.0f / H_DIM) + RMS_EPS);
  __syncthreads();
  const float sc = sscale;

  float4 w4 = ((const float4*)nw)[tid];
  float x0 = v.x * sc * w4.x, x1 = v.y * sc * w4.y;
  float x2 = v.z * sc * w4.z, x3 = v.w * sc * w4.w;
  ushort4 o;
  o.x = f2bf(x0); o.y = f2bf(x1); o.z = f2bf(x2); o.w = f2bf(x3);
  ((ushort4*)(xn + (size_t)t * H_DIM))[tid] = o;

  float part[E_NUM];
#pragma unroll
  for (int e = 0; e < E_NUM; e++) {
    float4 r4 = ((const float4*)(rw + e * H_DIM))[tid];
    part[e] = x0 * r4.x + x1 * r4.y + x2 * r4.z + x3 * r4.w;
  }
  for (int e = 0; e < E_NUM; e++) {
    __syncthreads();
    red[tid] = part[e];
    __syncthreads();
    for (int s = 128; s > 0; s >>= 1) {
      if (tid < s) red[tid] += red[tid + s];
      __syncthreads();
    }
    if (tid == 0) logits[e] = red[0] + rb[e];
  }
  __syncthreads();
  if (tid == 0) {
    float mx = logits[0];
    for (int e = 1; e < E_NUM; e++) mx = fmaxf(mx, logits[e]);
    float se = 0.f, pe[E_NUM];
    for (int e = 0; e < E_NUM; e++) { pe[e] = expf(logits[e] - mx); se += pe[e]; }
    float inv = 1.0f / se;
    for (int e = 0; e < E_NUM; e++)
      probsT[(size_t)e * T_NUM + t] = pe[e] * inv;
  }
}

// ---------------------------------------------------------------- per-expert top-CAP (radix select)
__global__ __launch_bounds__(1024)
void k_select(const float* __restrict__ probsT, int* __restrict__ idx,
              float* __restrict__ gates) {
  const int e = blockIdx.x;
  const int tid = threadIdx.x;
  const float* p = probsT + (size_t)e * T_NUM;

  __shared__ unsigned int hist[256];
  __shared__ unsigned int sh_prefix, sh_k, s_gt, s_eq;
  if (tid == 0) { sh_prefix = 0u; sh_k = CAP; s_gt = 0u; s_eq = 0u; }
  __syncthreads();

  for (int shift = 24; shift >= 0; shift -= 8) {
    if (tid < 256) hist[tid] = 0u;
    __syncthreads();
    unsigned int mask_hi = (shift == 24) ? 0u : (0xFFFFFFFFu << (shift + 8));
    unsigned int pref = sh_prefix;
    for (int i = tid; i < T_NUM; i += 1024) {
      unsigned int key = __float_as_uint(p[i]);   // probs >= 0: bit order == value order
      if ((key & mask_hi) == (pref & mask_hi))
        atomicAdd(&hist[(key >> shift) & 0xFFu], 1u);
    }
    __syncthreads();
    if (tid == 0) {
      unsigned int k = sh_k, cum = 0u; int chosen = 0;
      for (int b = 255; b >= 0; b--) {
        unsigned int c = hist[b];
        if (k <= cum + c) { chosen = b; break; }
        cum += c;
      }
      sh_prefix = pref | ((unsigned int)chosen << shift);
      sh_k = k - cum;
    }
    __syncthreads();
  }

  const unsigned int K = sh_prefix;      // exact key of CAP-th largest
  const unsigned int need = sh_k;        // how many ==K to take
  const unsigned int base_eq = CAP - need;
  for (int i = tid; i < T_NUM; i += 1024) {
    float pv = p[i];
    unsigned int key = __float_as_uint(pv);
    if (key > K) {
      unsigned int s = atomicAdd(&s_gt, 1u);
      if (s < base_eq) { idx[e * CAP + s] = i; gates[e * CAP + s] = pv; }
    } else if (key == K) {
      unsigned int q = atomicAdd(&s_eq, 1u);
      if (q < need) { idx[e * CAP + base_eq + q] = i; gates[e * CAP + base_eq + q] = pv; }
    }
  }
}

// ---------------------------------------------------------------- gate sums
__global__ void k_gate_sums(const int* __restrict__ idx, const float* __restrict__ gates,
                            float* __restrict__ gsum) {
  int i = blockIdx.x * blockDim.x + threadIdx.x;
  if (i < E_NUM * CAP) atomicAdd(&gsum[idx[i]], gates[i]);
}

// ================================================================ GEMM core
// Block tile 64(M) x 256(N), BK=32, 256 threads = 8 waves (2x4), per-wave
// 32x64 = 2x4 grid of 16x16 WMMA tiles (8 WMMA per 12 ds_load_b128).
// Single LDS buffer + two barriers: the scheduler pipelines global loads
// across the barrier without spilling (explicit reg-staging spilled).
#define BM 64
#define BN 256
#define BK 32
#define LDP 40   // padded LDS row stride (bf16 elements), 16B-aligned

// ---------------------------------------------------------------- fc1: hmid = GELU(xe @ W1^T + b1)
__global__ __launch_bounds__(256)
void k_fc1(const unsigned short* __restrict__ xn, const int* __restrict__ idx,
           const unsigned short* __restrict__ w, const float* __restrict__ bias,
           unsigned short* __restrict__ hmid) {
  const int e  = blockIdx.z;
  const int m0 = blockIdx.y * BM;
  const int n0 = blockIdx.x * BN;
  const int tid = threadIdx.x;

  __shared__ unsigned short As[BM][LDP];
  __shared__ unsigned short Bs[BN][LDP];
  __shared__ int rowidx[BM];

  if (tid < BM) rowidx[tid] = idx[e * CAP + m0 + tid];
  __syncthreads();

  const int ar = tid >> 2, ac = (tid & 3) * 8;   // A: 8 bf16 per thread
  const int brow = tid;                           // B: one 32-elem row per thread
  const unsigned short* abase = xn + (size_t)rowidx[ar] * H_DIM;
  const unsigned short* wbase = w + (size_t)(e * FF_DIM + n0 + brow) * H_DIM;

  const int wv = tid >> 5, lane = tid & 31;
  const int wm = (wv & 1) * 32, wn = (wv >> 1) * 64;
  const int lrow = lane & 15, lhi = lane >> 4;

  v8f acc[2][4] = {};

  for (int kk = 0; kk < H_DIM; kk += BK) {
    *(uint4*)&As[ar][ac] = *(const uint4*)(abase + kk + ac);
#pragma unroll
    for (int j = 0; j < 4; j++)
      *(uint4*)&Bs[brow][j * 8] = *(const uint4*)(wbase + kk + j * 8);
    if (kk + BK < H_DIM) {
      __builtin_prefetch(wbase + kk + BK, 0, 3);        // GL2 prefetch of next slab
      __builtin_prefetch(abase + kk + BK + ac, 0, 3);
    }
    __syncthreads();

    ABFrag a[2], b[4];
#pragma unroll
    for (int mi = 0; mi < 2; mi++) {
      const f4v* ap = (const f4v*)&As[wm + mi * 16 + lrow][0];
      a[mi].f[0] = ap[lhi];          // K = lhi*8 .. +7
      a[mi].f[1] = ap[2 + lhi];      // K = 16 + lhi*8 .. +7
    }
#pragma unroll
    for (int ni = 0; ni < 4; ni++) {
      const f4v* bp = (const f4v*)&Bs[wn + ni * 16 + lrow][0];
      b[ni].f[0] = bp[lhi];          // K = lhi*8 .. +7
      b[ni].f[1] = bp[2 + lhi];      // K = 16 + lhi*8 .. +7
    }
#pragma unroll
    for (int mi = 0; mi < 2; mi++)
#pragma unroll
      for (int ni = 0; ni < 4; ni++)
        acc[mi][ni] = __builtin_amdgcn_wmma_f32_16x16x32_bf16(
            false, a[mi].v, false, b[ni].v, (short)0, acc[mi][ni], false, false);
    __syncthreads();
  }

#pragma unroll
  for (int mi = 0; mi < 2; mi++) {
#pragma unroll
    for (int ni = 0; ni < 4; ni++) {
      int n = n0 + wn + ni * 16 + lrow;
      float bval = bias[e * FF_DIM + n];
#pragma unroll
      for (int r = 0; r < 8; r++) {
        int m = m0 + wm + mi * 16 + lhi * 8 + r;
        float v = acc[mi][ni][r] + bval;
        hmid[((size_t)(e * CAP + m)) * FF_DIM + n] = f2bf(gelu_exact(v));
      }
    }
  }
}

// ---------------------------------------------------------------- fc2: out += gate * (hmid @ W2^T + b2)
__global__ __launch_bounds__(256)
void k_fc2(const unsigned short* __restrict__ hmid, const unsigned short* __restrict__ w,
           const float* __restrict__ bias, const int* __restrict__ idx,
           const float* __restrict__ gates, float* __restrict__ out) {
  const int e  = blockIdx.z;
  const int m0 = blockIdx.y * BM;
  const int n0 = blockIdx.x * BN;
  const int tid = threadIdx.x;

  __shared__ unsigned short As[BM][LDP];
  __shared__ unsigned short Bs[BN][LDP];

  const int ar = tid >> 2, ac = (tid & 3) * 8;
  const int brow = tid;
  const unsigned short* abase = hmid + (size_t)(e * CAP + m0 + ar) * FF_DIM;
  const unsigned short* wbase = w + (size_t)(e * H_DIM + n0 + brow) * FF_DIM;

  const int wv = tid >> 5, lane = tid & 31;
  const int wm = (wv & 1) * 32, wn = (wv >> 1) * 64;
  const int lrow = lane & 15, lhi = lane >> 4;

  v8f acc[2][4] = {};

  for (int kk = 0; kk < FF_DIM; kk += BK) {
    *(uint4*)&As[ar][ac] = *(const uint4*)(abase + kk + ac);
#pragma unroll
    for (int j = 0; j < 4; j++)
      *(uint4*)&Bs[brow][j * 8] = *(const uint4*)(wbase + kk + j * 8);
    if (kk + BK < FF_DIM) {
      __builtin_prefetch(wbase + kk + BK, 0, 3);
      __builtin_prefetch(abase + kk + BK + ac, 0, 3);
    }
    __syncthreads();

    ABFrag a[2], b[4];
#pragma unroll
    for (int mi = 0; mi < 2; mi++) {
      const f4v* ap = (const f4v*)&As[wm + mi * 16 + lrow][0];
      a[mi].f[0] = ap[lhi];
      a[mi].f[1] = ap[2 + lhi];
    }
#pragma unroll
    for (int ni = 0; ni < 4; ni++) {
      const f4v* bp = (const f4v*)&Bs[wn + ni * 16 + lrow][0];
      b[ni].f[0] = bp[lhi];
      b[ni].f[1] = bp[2 + lhi];
    }
#pragma unroll
    for (int mi = 0; mi < 2; mi++)
#pragma unroll
      for (int ni = 0; ni < 4; ni++)
        acc[mi][ni] = __builtin_amdgcn_wmma_f32_16x16x32_bf16(
            false, a[mi].v, false, b[ni].v, (short)0, acc[mi][ni], false, false);
    __syncthreads();
  }

#pragma unroll
  for (int mi = 0; mi < 2; mi++) {
#pragma unroll
    for (int r = 0; r < 8; r++) {
      int m = m0 + wm + mi * 16 + lhi * 8 + r;
      int tok = idx[e * CAP + m];
      float g = gates[e * CAP + m];
#pragma unroll
      for (int ni = 0; ni < 4; ni++) {
        int n = n0 + wn + ni * 16 + lrow;
        float v = acc[mi][ni][r] + bias[e * H_DIM + n];
        atomicAdd(&out[(size_t)tok * H_DIM + n], v * g);
      }
    }
  }
}

// ---------------------------------------------------------------- out /= max(gate_sum, 1e-9)
__global__ void k_normalize(float* __restrict__ out, const float* __restrict__ gsum) {
  size_t i = (size_t)blockIdx.x * blockDim.x + threadIdx.x;  // over T*H/4
  float4 v = ((float4*)out)[i];
  int t = (int)(i >> 8);                                     // H/4 = 256 vec4 per token
  float inv = 1.0f / fmaxf(gsum[t], 1e-9f);
  v.x *= inv; v.y *= inv; v.z *= inv; v.w *= inv;
  ((float4*)out)[i] = v;
}

// ---------------------------------------------------------------- launch
extern "C" void kernel_launch(void* const* d_in, const int* in_sizes, int n_in,
                              void* d_out, int out_size, void* d_ws, size_t ws_size,
                              hipStream_t stream) {
  const float* x   = (const float*)d_in[0];
  const float* nw  = (const float*)d_in[1];
  const float* rw  = (const float*)d_in[2];
  const float* rb  = (const float*)d_in[3];
  const float* f1w = (const float*)d_in[4];
  const float* f1b = (const float*)d_in[5];
  const float* f2w = (const float*)d_in[6];
  const float* f2b = (const float*)d_in[7];
  float* out = (float*)d_out;

  char* ws = (char*)d_ws;
  // workspace layout (bytes)
  const size_t OFF_XN    = 0;                                              // 32 MB
  const size_t OFF_F1WB  = OFF_XN    + (size_t)T_NUM * H_DIM * 2;          // 64 MB
  const size_t OFF_F2WB  = OFF_F1WB  + (size_t)E_NUM * FF_DIM * H_DIM * 2; // 64 MB
  const size_t OFF_HMID  = OFF_F2WB  + (size_t)E_NUM * H_DIM * FF_DIM * 2; // 128 MB
  const size_t OFF_PROBS = OFF_HMID  + (size_t)E_NUM * CAP * FF_DIM * 2;   // 512 KB
  const size_t OFF_IDX   = OFF_PROBS + (size_t)E_NUM * T_NUM * 4;
  const size_t OFF_GATES = OFF_IDX   + (size_t)E_NUM * CAP * 4;
  const size_t OFF_GSUM  = OFF_GATES + (size_t)E_NUM * CAP * 4;

  unsigned short* xn     = (unsigned short*)(ws + OFF_XN);
  unsigned short* f1wb   = (unsigned short*)(ws + OFF_F1WB);
  unsigned short* f2wb   = (unsigned short*)(ws + OFF_F2WB);
  unsigned short* hmid   = (unsigned short*)(ws + OFF_HMID);
  float*          probsT = (float*)(ws + OFF_PROBS);
  int*            idxb   = (int*)(ws + OFF_IDX);
  float*          gatesb = (float*)(ws + OFF_GATES);
  float*          gsum   = (float*)(ws + OFF_GSUM);

  hipMemsetAsync(out,  0, (size_t)T_NUM * H_DIM * sizeof(float), stream);
  hipMemsetAsync(gsum, 0, (size_t)T_NUM * sizeof(float), stream);

  const int nW = E_NUM * FF_DIM * H_DIM;  // 33.5M per weight tensor
  k_f32_to_bf16x4<<<4096, 256, 0, stream>>>((const float4*)f1w, (ushort4*)f1wb, nW / 4);
  k_f32_to_bf16x4<<<4096, 256, 0, stream>>>((const float4*)f2w, (ushort4*)f2wb, nW / 4);

  k_rmsnorm_router<<<T_NUM, 256, 0, stream>>>(x, nw, rw, rb, xn, probsT);
  k_select<<<E_NUM, 1024, 0, stream>>>(probsT, idxb, gatesb);
  k_gate_sums<<<(E_NUM * CAP) / 256, 256, 0, stream>>>(idxb, gatesb, gsum);

  dim3 g1(FF_DIM / BN, CAP / BM, E_NUM);
  k_fc1<<<g1, 256, 0, stream>>>(xn, idxb, f1wb, f1b, hmid);

  dim3 g2(H_DIM / BN, CAP / BM, E_NUM);
  k_fc2<<<g2, 256, 0, stream>>>(hmid, f2wb, f2b, idxb, gatesb, out);

  k_normalize<<<(T_NUM * H_DIM / 4) / 256, 256, 0, stream>>>(out, gsum);
}
